// MCAttEGNN_33621003993979
// MI455X (gfx1250) — matrix operand order repaired and checked by reading, not compiled, not run on hardware
//
#include <hip/hip_runtime.h>
#include <hip/hip_bf16.h>
#include <math.h>

// ---------------------------------------------------------------------------
// MCAttEGNN forward for gfx1250 (MI455X).  All dense layers run through
// v_wmma_f32_16x16x32_bf16 (16x16 output tiles, K stepped by 32, f32 acc).
// Edge-feature gathers use GLOBAL_LOAD_ASYNC_TO_LDS_B128 (ASYNCcnt path).
// ---------------------------------------------------------------------------

#define N_NODES 10000
#define N_EDGES 320000
#define HID 128
#define IN_NF 64
#define OUT_NF 64
#define CCH 4          // C
#define RADF 32        // 2*C*C
#define KVIN 160       // HID + 2*C*C
#define EM0IN 288      // 2*HID + 2*C*C
#define NM0IN 256      // 2*HID

typedef __bf16 bf16_t;
typedef bf16_t v16bf __attribute__((ext_vector_type(16)));
typedef float  v8f   __attribute__((ext_vector_type(8)));

struct alignas(16) U4 { unsigned int x[4]; };
union FragU { v16bf v; U4 q[2]; };

// ---- async memory->LDS copy (16B per lane), tracked by ASYNCcnt ------------
// the vdst VGPR supplies the LDS destination address;
// so the low 32 bits of a flat shared pointer are a valid LDS byte address.
__device__ __forceinline__ void async_copy_b128(void* lds_dst, const void* gsrc) {
    unsigned lds = (unsigned)(size_t)lds_dst;
    asm volatile("global_load_async_to_lds_b128 %0, %1, off"
                 :: "v"(lds), "v"(gsrc) : "memory");
}
__device__ __forceinline__ void async_wait_all() {
    asm volatile("s_wait_asynccnt 0x0" ::: "memory");
}

// ---- WMMA fragment loaders -------------------------------------------------
// A tile is row-major [16][stride] bf16 (rows = M).  ISA 16-bit A layout:
// lane L: m=L%16, h=L/16; elems 0..7 = A[m][k0+8h..], elems 8..15 = A[m][k0+16+8h..]
__device__ __forceinline__ v16bf load_a_frag(const bf16_t* base, int stride, int k0, int lane) {
    int m = lane & 15, hh = lane >> 4;
    const bf16_t* r = base + (size_t)m * stride + k0 + 8 * hh;
    FragU f;
    f.q[0] = *(const U4*)(r);
    f.q[1] = *(const U4*)(r + 16);
    return f.v;
}
// B is a weight stored transposed: Wt[dout][din] row-major.  ISA 16-bit B
// layout: lane L: n=L%16, h=L/16; elems 0..7 = B[k0+8h..][n], elems 8..15 = B[k0+16+8h..][n]
// = Wt[col0+n][k0+8h .. +7], Wt[col0+n][k0+16+8h .. +7].
__device__ __forceinline__ v16bf load_b_frag(const bf16_t* wt, int din, int col0, int k0, int lane) {
    int n = lane & 15, hh = lane >> 4;
    const bf16_t* r = wt + (size_t)(col0 + n) * din + k0 + 8 * hh;
    FragU f;
    f.q[0] = *(const U4*)(r);
    f.q[1] = *(const U4*)(r + 16);
    return f.v;
}

// One 16x16 output tile (col tile col0), K = din stepped by 32.
__device__ __forceinline__ v8f wmma_tile(const bf16_t* a, int a_stride,
                                         const bf16_t* wt, int din, int col0, int lane) {
    v8f acc = {};
    for (int k0 = 0; k0 < din; k0 += 32) {
        v16bf af = load_a_frag(a, a_stride, k0, lane);
        v16bf bfr = load_b_frag(wt, din, col0, k0, lane);
        acc = __builtin_amdgcn_wmma_f32_16x16x32_bf16(false, af, false, bfr,
                                                      (short)0, acc, false, false);
    }
    return acc;
}

// Hint the next stage's weight matrix toward the caches (global_prefetch_b8).
__device__ __forceinline__ void prefetch_weights(const bf16_t* wt, int elems, int lane) {
    // one cacheline probe per lane, strided across the matrix
    const char* p = (const char*)wt + (size_t)lane * 128;
    if ((size_t)lane * 128 < (size_t)elems * 2) __builtin_prefetch(p, 0, 1);
}

__device__ __forceinline__ float silu_f(float v) { return v / (1.0f + __expf(-v)); }

// Store a 16x16 tile into a row-major bf16 LDS buffer, + bias, optional silu.
// D layout: lane L: n=L%16, h=L/16; acc[j] -> row j+8h, col col0+n.
__device__ __forceinline__ void store_tile_bf16(bf16_t* out, int stride, int col0,
                                                v8f acc, const float* bias, bool act, int lane) {
    int n = lane & 15, hh = lane >> 4;
    float b = bias ? bias[col0 + n] : 0.0f;
#pragma unroll
    for (int j = 0; j < 8; ++j) {
        float v = acc[j] + b;
        if (act) v = silu_f(v);
        out[(size_t)(j + 8 * hh) * stride + col0 + n] = (bf16_t)v;
    }
}

__device__ __forceinline__ void atomicMaxF(float* addr, float val) {
    unsigned int* ua = (unsigned int*)addr;
    unsigned int old = *ua;
    while (__uint_as_float(old) < val) {
        unsigned int assumed = old;
        old = atomicCAS(ua, assumed, __float_as_uint(val));
        if (old == assumed) break;
    }
}

// ---- small utility kernels -------------------------------------------------
__global__ void fill_kernel(float* p, float v, int n) {
    int i = blockIdx.x * blockDim.x + threadIdx.x;
    if (i < n) p[i] = v;
}
__global__ void refresh_hb_kernel(const float* h, bf16_t* hb, int n) {
    int i = blockIdx.x * blockDim.x + threadIdx.x;
    if (i < n) hb[i] = (bf16_t)h[i];
}

// ---- weight prep: fp32 [din][dout] -> bf16 transposed [dout][din] ----------
__global__ void prep_wt_kernel(const float* w, bf16_t* wt, int din, int dout) {
    int i = blockIdx.x * blockDim.x + threadIdx.x;
    if (i >= din * dout) return;
    int k = i / dout, n = i % dout;
    wt[(size_t)n * din + k] = (bf16_t)w[(size_t)k * dout + n];
}
// cm1: [128][4] -> padded transposed [16][128], cols >= dout are zero
__global__ void prep_wt_pad_kernel(const float* w, bf16_t* wt, int din, int dout, int dpad) {
    int i = blockIdx.x * blockDim.x + threadIdx.x;
    if (i >= din * dpad) return;
    int k = i / dpad, n = i % dpad;
    wt[(size_t)n * din + k] = (bf16_t)((n < dout) ? w[(size_t)k * dout + n] : 0.0f);
}
// kv: [160][256] -> Wk_t[128][160] (even cols), Wv_t[128][160] (odd cols)
__global__ void prep_kv_kernel(const float* w, const float* b,
                               bf16_t* wkt, bf16_t* wvt, float* bk, float* bv) {
    int i = blockIdx.x * blockDim.x + threadIdx.x;
    if (i >= KVIN * HID) return;
    int k = i / HID, n = i % HID;
    wkt[(size_t)n * KVIN + k] = (bf16_t)w[(size_t)k * 2 * HID + 2 * n];
    wvt[(size_t)n * KVIN + k] = (bf16_t)w[(size_t)k * 2 * HID + 2 * n + 1];
    if (k == 0) { bk[n] = b[2 * n]; bv[n] = b[2 * n + 1]; }
}

// ---- lin_in: h[N][64] @ W[64][128] + b -> h fp32 + bf16 mirror -------------
__global__ void __launch_bounds__(256)
lin_in_kernel(const float* __restrict__ h_in, const bf16_t* __restrict__ wt,
              const float* __restrict__ bias, float* __restrict__ h, bf16_t* __restrict__ hb) {
    __shared__ __align__(16) bf16_t A[16 * IN_NF];
    int node0 = blockIdx.x * 16, tid = threadIdx.x;
    for (int i = tid; i < 16 * IN_NF; i += 256)
        A[i] = (bf16_t)h_in[(size_t)(node0 + i / IN_NF) * IN_NF + (i % IN_NF)];
    __syncthreads();
    int wave = tid >> 5, lane = tid & 31;
    v8f acc = wmma_tile(A, IN_NF, wt, IN_NF, wave * 16, lane);
    int n = lane & 15, hh = lane >> 4;
    float b = bias[wave * 16 + n];
#pragma unroll
    for (int j = 0; j < 8; ++j) {
        float v = acc[j] + b;
        size_t idx = (size_t)(node0 + j + 8 * hh) * HID + wave * 16 + n;
        h[idx] = v; hb[idx] = (bf16_t)v;
    }
}

// ---- lin_out: hb[N][128] @ W[128][64] + b -> d_out fp32 --------------------
__global__ void lin_out_kernel(const bf16_t* __restrict__ hb, const bf16_t* __restrict__ wt,
                               const float* __restrict__ bias, float* __restrict__ out) {
    int node0 = blockIdx.x * 16, tid = threadIdx.x;
    int wave = tid >> 5, lane = tid & 31;            // 4 waves, 4 col tiles
    v8f acc = wmma_tile(hb + (size_t)node0 * HID, HID, wt, HID, wave * 16, lane);
    int n = lane & 15, hh = lane >> 4;
    float b = bias[wave * 16 + n];
#pragma unroll
    for (int j = 0; j < 8; ++j)
        out[(size_t)(node0 + j + 8 * hh) * OUT_NF + wave * 16 + n] = acc[j] + b;
}

// ---- radial pass 1: per-edge cd / raw radial+dist + global square sums -----
__global__ void __launch_bounds__(256)
radial_kernel(const float* __restrict__ coord, const int* __restrict__ row,
              const int* __restrict__ col, int nE, float* __restrict__ cd_out,
              float* __restrict__ raw, float* __restrict__ sums, float* cnt) {
    __shared__ float ls[RADF];
    int tid = threadIdx.x;
    if (tid < RADF) ls[tid] = 0.0f;
    __syncthreads();
    int e = blockIdx.x * blockDim.x + tid;
    if (e < nE) {
        int r = row[e], c = col[e];
        float cr[12], cc[12], cd[12];
#pragma unroll
        for (int j = 0; j < 12; ++j) {
            cr[j] = coord[(size_t)r * 12 + j];
            cc[j] = coord[(size_t)c * 12 + j];
            cd[j] = cr[j] - cc[j];
            cd_out[(size_t)e * 12 + j] = cd[j];
        }
        float sq[RADF];
#pragma unroll
        for (int ci = 0; ci < 4; ++ci) {
#pragma unroll
            for (int fi = 0; fi < 4; ++fi) {
                float rv = cd[ci*3]*cd[fi*3] + cd[ci*3+1]*cd[fi*3+1] + cd[ci*3+2]*cd[fi*3+2];
                float dx = cr[ci*3]   - cc[fi*3];
                float dy = cr[ci*3+1] - cc[fi*3+1];
                float dz = cr[ci*3+2] - cc[fi*3+2];
                float dv = sqrtf(fmaxf(dx*dx + dy*dy + dz*dz, 1e-24f));
                raw[(size_t)e * RADF + ci*8 + fi]     = rv;
                raw[(size_t)e * RADF + ci*8 + 4 + fi] = dv;
                sq[ci*8 + fi] = rv * rv;
                sq[ci*8 + 4 + fi] = dv * dv;
            }
        }
        if (cnt) atomicAdd(&cnt[r], 1.0f);
#pragma unroll
        for (int j = 0; j < RADF; ++j) atomicAdd(&ls[j], sq[j]);
    }
    __syncthreads();
    if (tid < RADF) atomicAdd(&sums[tid], ls[tid]);
}

// ---- radial pass 2: normalize -> bf16 --------------------------------------
__global__ void normalize_rad_kernel(const float* __restrict__ raw, const float* __restrict__ sums,
                                     bf16_t* __restrict__ rad, int nE) {
    size_t i = (size_t)blockIdx.x * blockDim.x + threadIdx.x;
    if (i >= (size_t)nE * RADF) return;
    int j = (int)(i & 31);
    float inv = 1.0f / fmaxf(sqrtf(sums[j]), 1e-12f);
    rad[i] = (bf16_t)(raw[i] * inv);
}

// ---- GCL edge MLP: em0->em1->cm0->cm1, scatter agg / coord_sum -------------
__global__ void __launch_bounds__(256)
gcl_edge_kernel(const bf16_t* __restrict__ hb, const bf16_t* __restrict__ rad,
                const float* __restrict__ cd, const int* __restrict__ row,
                const int* __restrict__ col,
                const bf16_t* em0w, const float* em0b, const bf16_t* em1w, const float* em1b,
                const bf16_t* cm0w, const float* cm0b, const bf16_t* cm1w,
                float* __restrict__ agg, float* __restrict__ coord_sum) {
    __shared__ __align__(16) bf16_t A0[16 * EM0IN];
    __shared__ __align__(16) bf16_t T1[16 * HID];
    __shared__ __align__(16) bf16_t T2[16 * HID];
    __shared__ float CM[16 * 16];
    __shared__ int rws[16], cls[16];
    int e0 = blockIdx.x * 16, tid = threadIdx.x;
    if (tid < 16) { rws[tid] = row[e0 + tid]; cls[tid] = col[e0 + tid]; }
    __syncthreads();
    int wave = tid >> 5, lane = tid & 31;
    // gather [h[row] | h[col] | rad] as async 16B chunks: 36 chunks per edge
    // (16 h_row + 16 h_col + 4 rad), 576 total over 256 threads.
    for (int i = tid; i < 16 * 36; i += 256) {
        int e = i / 36, j = i % 36;
        const bf16_t* src;
        if (j < 16)      src = hb  + (size_t)rws[e] * HID + j * 8;
        else if (j < 32) src = hb  + (size_t)cls[e] * HID + (j - 16) * 8;
        else             src = rad + (size_t)(e0 + e) * RADF + (j - 32) * 8;
        async_copy_b128(&A0[(size_t)e * EM0IN + j * 8], src);
    }
    prefetch_weights(em0w, EM0IN * HID, tid);
    async_wait_all();
    __syncthreads();
    v8f acc = wmma_tile(A0, EM0IN, em0w, EM0IN, wave * 16, lane);
    prefetch_weights(em1w, HID * HID, tid);
    store_tile_bf16(T1, HID, wave * 16, acc, em0b, true, lane);
    __syncthreads();
    acc = wmma_tile(T1, HID, em1w, HID, wave * 16, lane);
    prefetch_weights(cm0w, HID * HID, tid);
    store_tile_bf16(T2, HID, wave * 16, acc, em1b, true, lane);      // T2 = ef
    __syncthreads();
    acc = wmma_tile(T2, HID, cm0w, HID, wave * 16, lane);
    store_tile_bf16(T1, HID, wave * 16, acc, cm0b, true, lane);      // T1 = silu(cm0(ef))
    __syncthreads();
    if (wave == 0) {
        v8f c2 = wmma_tile(T1, HID, cm1w, HID, 0, lane);             // cm1 padded to 16 cols
        int n = lane & 15, hh = lane >> 4;
#pragma unroll
        for (int j = 0; j < 8; ++j) CM[(j + 8 * hh) * 16 + n] = c2[j];
    }
    __syncthreads();
    // agg[row] += ef
    for (int i = tid; i < 16 * HID; i += 256) {
        int e = i >> 7, j = i & 127;
        atomicAdd(&agg[(size_t)rws[e] * HID + j], (float)T2[(size_t)e * HID + j]);
    }
    // coord_sum[row] += cd * cm
    if (tid < 192) {
        int e = tid / 12, k = tid % 12, c = k / 3;
        float t = cd[(size_t)(e0 + e) * 12 + k] * CM[e * 16 + c];
        atomicAdd(&coord_sum[(size_t)rws[e] * 12 + k], t);
    }
}

// ---- GCL node update: nm0->nm1 residual + coord mean update ----------------
__global__ void __launch_bounds__(256)
gcl_node_kernel(float* __restrict__ h, bf16_t* __restrict__ hb, const float* __restrict__ agg,
                float* __restrict__ coord, const float* __restrict__ coord_sum,
                const float* __restrict__ cnt,
                const bf16_t* nm0w, const float* nm0b, const bf16_t* nm1w, const float* nm1b) {
    __shared__ __align__(16) bf16_t A[16 * NM0IN];
    __shared__ __align__(16) bf16_t T[16 * HID];
    int n0 = blockIdx.x * 16, tid = threadIdx.x;
    for (int i = tid; i < 16 * NM0IN; i += 256) {
        int r = i >> 8, j = i & 255;
        float v = (j < HID) ? h[(size_t)(n0 + r) * HID + j]
                            : agg[(size_t)(n0 + r) * HID + (j - HID)];
        A[i] = (bf16_t)v;
    }
    prefetch_weights(nm0w, NM0IN * HID, tid);
    __syncthreads();
    int wave = tid >> 5, lane = tid & 31;
    v8f acc = wmma_tile(A, NM0IN, nm0w, NM0IN, wave * 16, lane);
    prefetch_weights(nm1w, HID * HID, tid);
    store_tile_bf16(T, HID, wave * 16, acc, nm0b, true, lane);
    __syncthreads();
    acc = wmma_tile(T, HID, nm1w, HID, wave * 16, lane);
    int n = lane & 15, hh = lane >> 4;
    float b = nm1b[wave * 16 + n];
#pragma unroll
    for (int j = 0; j < 8; ++j) {
        size_t idx = (size_t)(n0 + j + 8 * hh) * HID + wave * 16 + n;
        float v = h[idx] + acc[j] + b;
        h[idx] = v; hb[idx] = (bf16_t)v;
    }
    if (tid < 192) {
        int r = tid / 12, j = tid % 12;
        float cv = fmaxf(cnt[n0 + r], 1.0f);
        coord[(size_t)(n0 + r) * 12 + j] += coord_sum[(size_t)(n0 + r) * 12 + j] / cv;
    }
}

// ---- ATT pass 1: q,k,v GEMMs, score = q.k, seg max -------------------------
__global__ void __launch_bounds__(256)
att_qkv_kernel(const bf16_t* __restrict__ hb, const bf16_t* __restrict__ rad,
               const int* __restrict__ row, const int* __restrict__ col,
               const bf16_t* qw, const float* qb, const bf16_t* wkt, const float* bk,
               const bf16_t* wvt, const float* bv,
               bf16_t* __restrict__ v_out, float* __restrict__ score, float* __restrict__ seg_max) {
    __shared__ __align__(16) bf16_t AQ[16 * HID];
    __shared__ __align__(16) bf16_t AKV[16 * KVIN];
    __shared__ float Q[16 * HID];
    __shared__ float K[16 * HID];
    __shared__ float SC[16];
    __shared__ int rws[16], cls[16];
    int e0 = blockIdx.x * 16, tid = threadIdx.x;
    if (tid < 16) { rws[tid] = row[e0 + tid]; cls[tid] = col[e0 + tid]; SC[tid] = 0.0f; }
    __syncthreads();
    // AQ: 16 chunks/edge = 256 chunks; AKV: 4 rad + 16 h_col = 20/edge = 320.
    for (int i = tid; i < 16 * 16; i += 256)
        async_copy_b128(&AQ[(size_t)(i >> 4) * HID + (i & 15) * 8],
                        hb + (size_t)rws[i >> 4] * HID + (i & 15) * 8);
    for (int i = tid; i < 16 * 20; i += 256) {
        int e = i / 20, j = i % 20;
        const bf16_t* src = (j < 4) ? rad + (size_t)(e0 + e) * RADF + j * 8
                                    : hb  + (size_t)cls[e] * HID + (j - 4) * 8;
        async_copy_b128(&AKV[(size_t)e * KVIN + j * 8], src);
    }
    prefetch_weights(qw, HID * HID, tid);
    async_wait_all();
    __syncthreads();
    int wave = tid >> 5, lane = tid & 31;
    int n = lane & 15, hh = lane >> 4;
    v8f acc = wmma_tile(AQ, HID, qw, HID, wave * 16, lane);
    prefetch_weights(wkt, KVIN * HID, tid);
    {
        float b = qb[wave * 16 + n];
#pragma unroll
        for (int j = 0; j < 8; ++j) Q[(j + 8 * hh) * HID + wave * 16 + n] = acc[j] + b;
    }
    acc = wmma_tile(AKV, KVIN, wkt, KVIN, wave * 16, lane);
    prefetch_weights(wvt, KVIN * HID, tid);
    {
        float b = bk[wave * 16 + n];
#pragma unroll
        for (int j = 0; j < 8; ++j) K[(j + 8 * hh) * HID + wave * 16 + n] = acc[j] + b;
    }
    acc = wmma_tile(AKV, KVIN, wvt, KVIN, wave * 16, lane);
    {
        float b = bv[wave * 16 + n];
#pragma unroll
        for (int j = 0; j < 8; ++j)
            v_out[(size_t)(e0 + j + 8 * hh) * HID + wave * 16 + n] = (bf16_t)(acc[j] + b);
    }
    __syncthreads();
    {   // score = sum(q*k): 16 threads per edge, 8 elems each
        int e = tid >> 4, j0 = (tid & 15) * 8;
        float s = 0.0f;
#pragma unroll
        for (int j = 0; j < 8; ++j) s += Q[e * HID + j0 + j] * K[e * HID + j0 + j];
        atomicAdd(&SC[e], s);
    }
    __syncthreads();
    if (tid < 16) {
        float s = SC[tid];
        score[e0 + tid] = s;
        atomicMaxF(&seg_max[rws[tid]], s);
    }
}

// ---- ATT pass 2: exp(score - max) + seg sum --------------------------------
__global__ void att_exp_kernel(float* __restrict__ score, const float* __restrict__ seg_max,
                               const int* __restrict__ row, float* __restrict__ seg_sum, int nE) {
    int e = blockIdx.x * blockDim.x + threadIdx.x;
    if (e >= nE) return;
    float m = seg_max[row[e]];
    if (!(m > -1e37f)) m = 0.0f;                 // where(isfinite(m), m, 0)
    float ev = __expf(score[e] - m);
    score[e] = ev;
    atomicAdd(&seg_sum[row[e]], ev);
}

// ---- ATT pass 3: alpha, h += alpha*v, coord += cd*alpha*cm -----------------
__global__ void __launch_bounds__(256)
att_apply_kernel(const bf16_t* __restrict__ v_in, const float* __restrict__ escore,
                 const float* __restrict__ seg_sum, const int* __restrict__ row,
                 const float* __restrict__ cd,
                 const bf16_t* cm0w, const float* cm0b, const bf16_t* cm1w,
                 float* __restrict__ h, float* __restrict__ coord) {
    __shared__ __align__(16) bf16_t T[16 * HID];
    __shared__ float CM[16 * 16];
    __shared__ float AL[16];
    __shared__ int rws[16];
    int e0 = blockIdx.x * 16, tid = threadIdx.x;
    if (tid < 16) {
        int r = row[e0 + tid];
        rws[tid] = r;
        AL[tid] = escore[e0 + tid] / seg_sum[r];
    }
    __syncthreads();
    int wave = tid >> 5, lane = tid & 31;
    v8f acc = wmma_tile(v_in + (size_t)e0 * HID, HID, cm0w, HID, wave * 16, lane);
    store_tile_bf16(T, HID, wave * 16, acc, cm0b, true, lane);
    __syncthreads();
    if (wave == 0) {
        v8f c2 = wmma_tile(T, HID, cm1w, HID, 0, lane);
        int n = lane & 15, hh = lane >> 4;
#pragma unroll
        for (int j = 0; j < 8; ++j) CM[(j + 8 * hh) * 16 + n] = c2[j];
    }
    __syncthreads();
    for (int i = tid; i < 16 * HID; i += 256) {
        int e = i >> 7, j = i & 127;
        atomicAdd(&h[(size_t)rws[e] * HID + j], AL[e] * (float)v_in[(size_t)(e0 + e) * HID + j]);
    }
    if (tid < 192) {
        int e = tid / 12, k = tid % 12, c = k / 3;
        float t = cd[(size_t)(e0 + e) * 12 + k] * AL[e] * CM[e * 16 + c];
        atomicAdd(&coord[(size_t)rws[e] * 12 + k], t);
    }
}

// ===========================================================================
extern "C" void kernel_launch(void* const* d_in, const int* in_sizes, int n_in,
                              void* d_out, int out_size, void* d_ws, size_t ws_size,
                              hipStream_t stream) {
    (void)in_sizes; (void)n_in; (void)out_size; (void)ws_size;
    const int N = N_NODES, E = N_EDGES;

    // -- input mapping (recursive dict-insertion order of setup_inputs()) ----
    const float* P[91];
    for (int i = 0; i < 91; ++i) P[i] = (const float*)d_in[i];
    const float* d_h0 = P[0];
    const float* d_x0 = P[1];
    // 2,3: lin_in.w/b ; 4,5: lin_out.w/b ; 6..16: out_layer ;
    // 17+11*i: gcl[i] ; 61+7*i: att[i] ; 89: ctx_edges ; 90: att_edges
    const int* ctx = (const int*)d_in[89];
    const int* att = (const int*)d_in[90];
    const int* ctx_row = ctx, * ctx_col = ctx + E;
    const int* att_row = att, * att_col = att + E;

    struct GclW { const float *em0w,*em0b,*em1w,*em1b,*nm0w,*nm0b,*nm1w,*nm1b,*cm0w,*cm0b,*cm1w; };
    struct AttW { const float *qw,*qb,*kvw,*kvb,*cm0w,*cm0b,*cm1w; };
    auto getGcl = [&](int b) {
        GclW g{P[b],P[b+1],P[b+2],P[b+3],P[b+4],P[b+5],P[b+6],P[b+7],P[b+8],P[b+9],P[b+10]};
        return g;
    };
    auto getAtt = [&](int b) {
        AttW a{P[b],P[b+1],P[b+2],P[b+3],P[b+4],P[b+5],P[b+6]};
        return a;
    };
    GclW gclP[5]; AttW attP[4];
    gclP[4] = getGcl(6);                               // out_layer
    for (int i = 0; i < 4; ++i) gclP[i] = getGcl(17 + 11 * i);
    for (int i = 0; i < 4; ++i) attP[i] = getAtt(61 + 7 * i);

    // -- workspace bump allocator -------------------------------------------
    char* wp = (char*)d_ws;
    auto alloc = [&](size_t bytes) -> void* {
        void* p = (void*)wp;
        wp += (bytes + 255) & ~(size_t)255;
        return p;
    };
    float*  h         = (float*) alloc((size_t)N * HID * 4);
    bf16_t* hb        = (bf16_t*)alloc((size_t)N * HID * 2);
    float*  coord     = (float*) alloc((size_t)N * 12 * 4);
    float*  cd        = (float*) alloc((size_t)E * 12 * 4);
    float*  raw       = (float*) alloc((size_t)E * RADF * 4);
    bf16_t* rad       = (bf16_t*)alloc((size_t)E * RADF * 2);
    float*  sums      = (float*) alloc(RADF * 4);
    float*  cnt       = (float*) alloc((size_t)N * 4);
    float*  agg       = (float*) alloc((size_t)N * HID * 4);
    float*  coord_sum = (float*) alloc((size_t)N * 12 * 4);
    float*  seg_max   = (float*) alloc((size_t)N * 4);
    float*  seg_sum   = (float*) alloc((size_t)N * 4);
    float*  score     = (float*) alloc((size_t)E * 4);
    bf16_t* vbuf      = (bf16_t*)alloc((size_t)E * HID * 2);

    // bf16 transposed weights
    struct GclB { bf16_t *em0,*em1,*nm0,*nm1,*cm0,*cm1; };
    struct AttB { bf16_t *q,*wk,*wv,*cm0,*cm1; float *bk,*bv; };
    bf16_t* lin_in_wt  = (bf16_t*)alloc((size_t)IN_NF * HID * 2);
    bf16_t* lin_out_wt = (bf16_t*)alloc((size_t)HID * OUT_NF * 2);
    GclB gclB[5]; AttB attB[4];
    for (int i = 0; i < 5; ++i) {
        gclB[i].em0 = (bf16_t*)alloc((size_t)EM0IN * HID * 2);
        gclB[i].em1 = (bf16_t*)alloc((size_t)HID * HID * 2);
        gclB[i].nm0 = (bf16_t*)alloc((size_t)NM0IN * HID * 2);
        gclB[i].nm1 = (bf16_t*)alloc((size_t)HID * HID * 2);
        gclB[i].cm0 = (bf16_t*)alloc((size_t)HID * HID * 2);
        gclB[i].cm1 = (bf16_t*)alloc((size_t)HID * 16 * 2);
    }
    for (int i = 0; i < 4; ++i) {
        attB[i].q   = (bf16_t*)alloc((size_t)HID * HID * 2);
        attB[i].wk  = (bf16_t*)alloc((size_t)KVIN * HID * 2);
        attB[i].wv  = (bf16_t*)alloc((size_t)KVIN * HID * 2);
        attB[i].cm0 = (bf16_t*)alloc((size_t)HID * HID * 2);
        attB[i].cm1 = (bf16_t*)alloc((size_t)HID * 16 * 2);
        attB[i].bk  = (float*)alloc(HID * 4);
        attB[i].bv  = (float*)alloc(HID * 4);
    }

    auto prep = [&](const float* w, bf16_t* wt, int din, int dout) {
        int n = din * dout;
        prep_wt_kernel<<<(n + 255) / 256, 256, 0, stream>>>(w, wt, din, dout);
    };
    prep(P[2], lin_in_wt, IN_NF, HID);
    prep(P[4], lin_out_wt, HID, OUT_NF);
    for (int i = 0; i < 5; ++i) {
        prep(gclP[i].em0w, gclB[i].em0, EM0IN, HID);
        prep(gclP[i].em1w, gclB[i].em1, HID, HID);
        prep(gclP[i].nm0w, gclB[i].nm0, NM0IN, HID);
        prep(gclP[i].nm1w, gclB[i].nm1, HID, HID);
        prep(gclP[i].cm0w, gclB[i].cm0, HID, HID);
        prep_wt_pad_kernel<<<(HID * 16 + 255) / 256, 256, 0, stream>>>(gclP[i].cm1w, gclB[i].cm1, HID, CCH, 16);
    }
    for (int i = 0; i < 4; ++i) {
        prep(attP[i].qw, attB[i].q, HID, HID);
        prep_kv_kernel<<<(KVIN * HID + 255) / 256, 256, 0, stream>>>(attP[i].kvw, attP[i].kvb,
                attB[i].wk, attB[i].wv, attB[i].bk, attB[i].bv);
        prep(attP[i].cm0w, attB[i].cm0, HID, HID);
        prep_wt_pad_kernel<<<(HID * 16 + 255) / 256, 256, 0, stream>>>(attP[i].cm1w, attB[i].cm1, HID, CCH, 16);
    }

    // -- forward -------------------------------------------------------------
    hipMemcpyAsync(coord, d_x0, (size_t)N * 12 * 4, hipMemcpyDeviceToDevice, stream);
    lin_in_kernel<<<N / 16, 256, 0, stream>>>(d_h0, lin_in_wt, P[3], h, hb);

    auto run_gcl = [&](const GclB& wb, const GclW& wf) {
        hipMemsetAsync(sums, 0, RADF * 4, stream);
        hipMemsetAsync(cnt, 0, (size_t)N * 4, stream);
        hipMemsetAsync(agg, 0, (size_t)N * HID * 4, stream);
        hipMemsetAsync(coord_sum, 0, (size_t)N * 12 * 4, stream);
        radial_kernel<<<E / 256, 256, 0, stream>>>(coord, ctx_row, ctx_col, E, cd, raw, sums, cnt);
        normalize_rad_kernel<<<E * RADF / 256, 256, 0, stream>>>(raw, sums, rad, E);
        gcl_edge_kernel<<<E / 16, 256, 0, stream>>>(hb, rad, cd, ctx_row, ctx_col,
                wb.em0, wf.em0b, wb.em1, wf.em1b, wb.cm0, wf.cm0b, wb.cm1, agg, coord_sum);
        gcl_node_kernel<<<N / 16, 256, 0, stream>>>(h, hb, agg, coord, coord_sum, cnt,
                wb.nm0, wf.nm0b, wb.nm1, wf.nm1b);
    };
    auto run_att = [&](const AttB& wb, const AttW& wf) {
        hipMemsetAsync(sums, 0, RADF * 4, stream);
        hipMemsetAsync(seg_sum, 0, (size_t)N * 4, stream);
        fill_kernel<<<(N + 255) / 256, 256, 0, stream>>>(seg_max, -__builtin_huge_valf(), N);
        radial_kernel<<<E / 256, 256, 0, stream>>>(coord, att_row, att_col, E, cd, raw, sums, nullptr);
        normalize_rad_kernel<<<E * RADF / 256, 256, 0, stream>>>(raw, sums, rad, E);
        att_qkv_kernel<<<E / 16, 256, 0, stream>>>(hb, rad, att_row, att_col,
                wb.q, wf.qb, wb.wk, wb.bk, wb.wv, wb.bv, vbuf, score, seg_max);
        att_exp_kernel<<<E / 256, 256, 0, stream>>>(score, seg_max, att_row, seg_sum, E);
        att_apply_kernel<<<E / 16, 256, 0, stream>>>(vbuf, score, seg_sum, att_row, cd,
                wb.cm0, wf.cm0b, wb.cm1, h, coord);
        refresh_hb_kernel<<<(N * HID) / 256, 256, 0, stream>>>(h, hb, N * HID);
    };

    for (int i = 0; i < 4; ++i) {
        run_gcl(gclB[i], gclP[i]);
        run_att(attB[i], attP[i]);
    }
    run_gcl(gclB[4], gclP[4]);                                     // out_layer

    lin_out_kernel<<<N / 16, 128, 0, stream>>>(hb, lin_out_wt, P[5], (float*)d_out);
    hipMemcpyAsync((float*)d_out + (size_t)N * OUT_NF, coord, (size_t)N * 12 * 4,
                   hipMemcpyDeviceToDevice, stream);
}
